// GNNLoRA_47021301956658
// MI455X (gfx1250) — matrix operand order repaired and checked by reading, not compiled, not run on hardware
//
#include <hip/hip_runtime.h>
#include <hip/hip_bf16.h>
#include <math.h>

// ---------------------------------------------------------------------------
// GAT + LoRA dual-layer pipeline for MI455X (gfx1250, wave32, WMMA).
//
// Dense GEMMs run on v_wmma_f32_16x16x32_bf16 (bf16 A/B, f32 accum).
// Edge passes (softmax max / sum / scatter) are L2-bandwidth bound and use
// float atomics (global_atomic_add_f32) -- the full working set (~100MB)
// fits in the 192MB L2 of the MI455X.  Layer-0 scatter fuses base+LoRA
// contributions in registers before the atomic (halves atomic traffic).
// ---------------------------------------------------------------------------

#define NODES 50000
#define EDGES 800000
#define NE2   (EDGES + NODES)   // edges + self loops
#define DIN   128
#define HIDC  128
#define OUTC  64
#define RANK  32

typedef __attribute__((ext_vector_type(16))) __bf16       v16bf;
typedef __attribute__((ext_vector_type(8)))  float        v8f;
typedef __attribute__((ext_vector_type(4)))  unsigned int u32x4;

// ---------------------------------------------------------------- conversions
__global__ void k_f32_to_bf16(const float* __restrict__ in,
                              __bf16* __restrict__ out, int n) {
  int i = blockIdx.x * blockDim.x + threadIdx.x;
  if (i < n) out[i] = (__bf16)in[i];
}

// weight convert + transpose: in [K,C] f32 row-major -> out [C,K] bf16
__global__ void k_wt_bf16(const float* __restrict__ in,
                          __bf16* __restrict__ out, int K, int C) {
  int i = blockIdx.x * blockDim.x + threadIdx.x;
  if (i >= K * C) return;
  int k = i / C, c = i - k * C;
  out[c * K + k] = (__bf16)in[i];
}

// ---------------------------------------------------------------- WMMA GEMM
// D[M,Nc] = A[M,K] x B[K,Nc], A row-major bf16, Bt = B transposed [Nc,K] bf16.
// One wave per 16x16 output tile; 4 waves (4 m-tiles) per block.
// A fragment (ISA 7.12.2, 16-bit A 16x32): lane half h, K = 32*kt + h*8+{0..7}
//   and 32*kt + 16 + h*8 + {0..7}.  B fragment: same K order as A.
__global__ void k_wmma_gemm(const __bf16* __restrict__ A,
                            const __bf16* __restrict__ Bt,
                            float* __restrict__ Cf,     // f32 out (or null)
                            __bf16* __restrict__ Cb,    // bf16 out (or null)
                            int M, int K, int Nc) {
  const int wave = threadIdx.x >> 5;
  const int lane = threadIdx.x & 31;
  const int m0 = (blockIdx.x * 4 + wave) * 16;
  const int n0 = blockIdx.y * 16;
  if (m0 >= M) return;                       // wave-uniform: EXEC stays all-1s
  const int half = lane >> 4;
  const int r    = lane & 15;

  const __bf16* Arow = A  + (size_t)(m0 + r) * K + half * 8;
  const __bf16* Bcol = Bt + (size_t)(n0 + r) * K + half * 8;

  v8f acc = {0.f, 0.f, 0.f, 0.f, 0.f, 0.f, 0.f, 0.f};
  for (int k0 = 0; k0 < K; k0 += 32) {
    v16bf fa, fb;
    ((u32x4*)&fa)[0] = *(const u32x4*)(Arow + k0);
    ((u32x4*)&fa)[1] = *(const u32x4*)(Arow + k0 + 16);
    ((u32x4*)&fb)[0] = *(const u32x4*)(Bcol + k0);
    ((u32x4*)&fb)[1] = *(const u32x4*)(Bcol + k0 + 16);
    acc = __builtin_amdgcn_wmma_f32_16x16x32_bf16(
        /*neg_a=*/false, fa, /*neg_b=*/false, fb,
        /*c_mod=*/(short)0, acc, /*reuse_a=*/false, /*reuse_b=*/false);
  }

  const int col = n0 + r;
  if (Cf) {
#pragma unroll
    for (int i = 0; i < 8; ++i) {
      int row = m0 + i + half * 8;           // C/D layout: lanes16-31 -> M+8
      Cf[(size_t)row * Nc + col] = acc[i];
    }
  } else {
#pragma unroll
    for (int i = 0; i < 8; ++i) {
      int row = m0 + i + half * 8;
      Cb[(size_t)row * Nc + col] = (__bf16)acc[i];
    }
  }
}

// ------------------------------------------------------- per-node attention
// alpha_src/dst for base+lora paths, plus init of softmax state.
__global__ void k_node_alpha(const float* __restrict__ hb,
                             const float* __restrict__ hl,
                             const float* __restrict__ att_sb,
                             const float* __restrict__ att_db,
                             const float* __restrict__ att_sl,
                             const float* __restrict__ att_dl,
                             float* asb, float* adb, float* asl, float* adl,
                             float* emaxb, float* emaxl,
                             float* denb, float* denl, int C) {
  __shared__ float red[4][128];
  const int n = blockIdx.x;
  const int c = threadIdx.x;
  float v0 = 0.f, v1 = 0.f, v2 = 0.f, v3 = 0.f;
  if (c < C) {
    float hbv = hb[(size_t)n * C + c];
    float hlv = hl[(size_t)n * C + c];
    v0 = hbv * att_sb[c]; v1 = hbv * att_db[c];
    v2 = hlv * att_sl[c]; v3 = hlv * att_dl[c];
  }
  red[0][c] = v0; red[1][c] = v1; red[2][c] = v2; red[3][c] = v3;
  __syncthreads();
  for (int s = 64; s > 0; s >>= 1) {
    if (c < s) {
      red[0][c] += red[0][c + s]; red[1][c] += red[1][c + s];
      red[2][c] += red[2][c + s]; red[3][c] += red[3][c + s];
    }
    __syncthreads();
  }
  if (c == 0) {
    asb[n] = red[0][0]; adb[n] = red[1][0];
    asl[n] = red[2][0]; adl[n] = red[3][0];
    emaxb[n] = -INFINITY; emaxl[n] = -INFINITY;
    denb[n] = 0.f;        denl[n] = 0.f;
  }
}

// ------------------------------------------------------------- edge helpers
__device__ __forceinline__ void edge_ends(const int* __restrict__ ei, int e,
                                          int& s, int& d) {
  if (e < EDGES) { s = ei[e]; d = ei[EDGES + e]; }
  else           { s = e - EDGES; d = s; }          // self loops
}
__device__ __forceinline__ float lrelu(float x) {
  return x > 0.f ? x : 0.2f * x;
}
__device__ __forceinline__ void atomicMaxF(float* addr, float val) {
  int* ia = (int*)addr;
  int old = __float_as_int(*addr);
  while (__int_as_float(old) < val) {
    int assumed = old;
    old = atomicCAS(ia, assumed, __float_as_int(val));
    if (old == assumed) break;
  }
}

// pass 1: segment max of leaky-relu'd logits (both paths)
__global__ void k_edge_max(const int* __restrict__ ei,
                           const float* __restrict__ asb, const float* __restrict__ adb,
                           const float* __restrict__ asl, const float* __restrict__ adl,
                           float* emaxb, float* emaxl) {
  int e = blockIdx.x * blockDim.x + threadIdx.x;
  if (e >= NE2) return;
  int s, d; edge_ends(ei, e, s, d);
  atomicMaxF(&emaxb[d], lrelu(asb[s] + adb[d]));
  atomicMaxF(&emaxl[d], lrelu(asl[s] + adl[d]));
}

// pass 2: ex = exp(e - max), segment sum (both paths)
__global__ void k_edge_exp(const int* __restrict__ ei,
                           const float* __restrict__ asb, const float* __restrict__ adb,
                           const float* __restrict__ asl, const float* __restrict__ adl,
                           const float* __restrict__ emaxb, const float* __restrict__ emaxl,
                           float* __restrict__ exb, float* __restrict__ exl,
                           float* denb, float* denl) {
  int e = blockIdx.x * blockDim.x + threadIdx.x;
  if (e >= NE2) return;
  int s, d; edge_ends(ei, e, s, d);
  float vb = __expf(lrelu(asb[s] + adb[d]) - emaxb[d]);
  float vl = __expf(lrelu(asl[s] + adl[d]) - emaxl[d]);
  exb[e] = vb; exl[e] = vl;
  atomicAdd(&denb[d], vb);
  atomicAdd(&denl[d], vl);
}

// pass 3: out[dst] += h[src] * alpha   (one thread per (edge, 4 channels)).
// When outB == outL (layer 0: both paths sum into x1) the two contributions
// are combined in registers first -> 4 atomics instead of 8 per thread.
__global__ void k_edge_scatter(const int* __restrict__ ei,
                               const float* __restrict__ hb,
                               const float* __restrict__ hl,
                               const float* __restrict__ exb,
                               const float* __restrict__ exl,
                               const float* __restrict__ denb,
                               const float* __restrict__ denl,
                               float* __restrict__ outB,
                               float* __restrict__ outL, int C) {
  const int cgc = C >> 2;
  int gid = blockIdx.x * blockDim.x + threadIdx.x;
  int e = gid / cgc;
  if (e >= NE2) return;
  int cg = (gid - e * cgc) << 2;
  int s, d; edge_ends(ei, e, s, d);
  float ab = exb[e] / (denb[d] + 1e-16f);
  float al = exl[e] / (denl[d] + 1e-16f);
  const float4 hbv = *(const float4*)(hb + (size_t)s * C + cg);
  const float4 hlv = *(const float4*)(hl + (size_t)s * C + cg);
  float* ob = outB + (size_t)d * C + cg;
  if (outB == outL) {                       // uniform branch (layer 0)
    atomicAdd(ob + 0, hbv.x * ab + hlv.x * al);
    atomicAdd(ob + 1, hbv.y * ab + hlv.y * al);
    atomicAdd(ob + 2, hbv.z * ab + hlv.z * al);
    atomicAdd(ob + 3, hbv.w * ab + hlv.w * al);
  } else {                                  // layer 1: separate e1 / e2
    float* ol = outL + (size_t)d * C + cg;
    atomicAdd(ob + 0, hbv.x * ab); atomicAdd(ob + 1, hbv.y * ab);
    atomicAdd(ob + 2, hbv.z * ab); atomicAdd(ob + 3, hbv.w * ab);
    atomicAdd(ol + 0, hlv.x * al); atomicAdd(ol + 1, hlv.y * al);
    atomicAdd(ol + 2, hlv.z * al); atomicAdd(ol + 3, hlv.w * al);
  }
}

// ----------------------------------------------------------------- epilogue
__global__ void k_init_bias(float* __restrict__ out,
                            const float* __restrict__ b1,
                            const float* __restrict__ b2, int n, int C) {
  int i = blockIdx.x * blockDim.x + threadIdx.x;
  if (i >= n) return;
  int c = i % C;
  float v = b1[c];
  if (b2) v += b2[c];
  out[i] = v;
}

__global__ void k_final_sum(const float* __restrict__ e1,
                            const float* __restrict__ e2,
                            float* __restrict__ out, int n) {
  int i = blockIdx.x * blockDim.x + threadIdx.x;
  if (i < n) out[i] = e1[i] + e2[i];
}

// ---------------------------------------------------------------------------
extern "C" void kernel_launch(void* const* d_in, const int* in_sizes, int n_in,
                              void* d_out, int out_size, void* d_ws, size_t ws_size,
                              hipStream_t stream) {
  // ---- inputs (setup_inputs order)
  const float* x       = (const float*)d_in[0];
  const int*   ei      = (const int*)  d_in[1];   // [2,E] edge_index
  const float* w0      = (const float*)d_in[2];
  const float* att_s0  = (const float*)d_in[3];
  const float* att_d0  = (const float*)d_in[4];
  const float* b0      = (const float*)d_in[5];
  const float* w1      = (const float*)d_in[6];
  const float* att_s1  = (const float*)d_in[7];
  const float* att_d1  = (const float*)d_in[8];
  const float* b1      = (const float*)d_in[9];
  const float* la0     = (const float*)d_in[10];
  const float* lb0     = (const float*)d_in[11];
  const float* latt_s0 = (const float*)d_in[12];
  const float* latt_d0 = (const float*)d_in[13];
  const float* lbias0  = (const float*)d_in[14];
  const float* la1     = (const float*)d_in[15];
  const float* lb1     = (const float*)d_in[16];
  const float* latt_s1 = (const float*)d_in[17];
  const float* latt_d1 = (const float*)d_in[18];
  const float* lbias1  = (const float*)d_in[19];
  float* out = (float*)d_out;

  // ---- workspace carve-up (256B aligned regions)
  size_t off = 0;
  auto carve = [&](size_t bytes) -> void* {
    void* p = (char*)d_ws + off;
    off += (bytes + 255) & ~(size_t)255;
    return p;
  };
  __bf16* xb   = (__bf16*)carve((size_t)NODES * DIN * 2);   // x / x1 in bf16
  float*  hb   = (float*) carve((size_t)NODES * HIDC * 4);  // base h
  float*  hl   = (float*) carve((size_t)NODES * HIDC * 4);  // lora h
  __bf16* t16  = (__bf16*)carve((size_t)NODES * RANK * 2);  // x @ A  (bf16)
  float*  x1   = (float*) carve((size_t)NODES * HIDC * 4);  // layer-0 output
  float*  asb  = (float*) carve((size_t)NODES * 4);
  float*  adb  = (float*) carve((size_t)NODES * 4);
  float*  asl  = (float*) carve((size_t)NODES * 4);
  float*  adl  = (float*) carve((size_t)NODES * 4);
  float*  emb  = (float*) carve((size_t)NODES * 4);
  float*  eml  = (float*) carve((size_t)NODES * 4);
  float*  denb = (float*) carve((size_t)NODES * 4);
  float*  denl = (float*) carve((size_t)NODES * 4);
  float*  exb  = (float*) carve((size_t)NE2 * 4);
  float*  exl  = (float*) carve((size_t)NE2 * 4);
  __bf16* w0t  = (__bf16*)carve((size_t)HIDC * DIN * 2);    // [C,K] transposed
  __bf16* la0t = (__bf16*)carve((size_t)RANK * DIN * 2);
  __bf16* lb0t = (__bf16*)carve((size_t)HIDC * RANK * 2);
  __bf16* w1t  = (__bf16*)carve((size_t)OUTC * HIDC * 2);
  __bf16* la1t = (__bf16*)carve((size_t)RANK * HIDC * 2);
  __bf16* lb1t = (__bf16*)carve((size_t)OUTC * RANK * 2);
  (void)ws_size; (void)n_in; (void)in_sizes; (void)out_size;

  const int TPB = 256;
  auto cdiv = [](long long a, long long b) { return (int)((a + b - 1) / b); };

  // ---- weight/activation conversion to bf16 (weights transposed to [C,K])
  k_f32_to_bf16<<<cdiv((long long)NODES * DIN, TPB), TPB, 0, stream>>>(x, xb, NODES * DIN);
  k_wt_bf16<<<cdiv(DIN * HIDC, TPB), TPB, 0, stream>>>(w0,  w0t,  DIN,  HIDC);
  k_wt_bf16<<<cdiv(DIN * RANK, TPB), TPB, 0, stream>>>(la0, la0t, DIN,  RANK);
  k_wt_bf16<<<cdiv(RANK * HIDC, TPB), TPB, 0, stream>>>(lb0, lb0t, RANK, HIDC);
  k_wt_bf16<<<cdiv(HIDC * OUTC, TPB), TPB, 0, stream>>>(w1,  w1t,  HIDC, OUTC);
  k_wt_bf16<<<cdiv(HIDC * RANK, TPB), TPB, 0, stream>>>(la1, la1t, HIDC, RANK);
  k_wt_bf16<<<cdiv(RANK * OUTC, TPB), TPB, 0, stream>>>(lb1, lb1t, RANK, OUTC);

  const int mblk = cdiv(NODES / 16, 4);  // 4 m-tiles (waves) per block

  // ================= layer 0 (C = 128) =================
  k_wmma_gemm<<<dim3(mblk, HIDC / 16), 128, 0, stream>>>(xb, w0t,  hb, nullptr, NODES, DIN, HIDC);
  k_wmma_gemm<<<dim3(mblk, RANK / 16), 128, 0, stream>>>(xb, la0t, nullptr, t16, NODES, DIN, RANK);
  k_wmma_gemm<<<dim3(mblk, HIDC / 16), 128, 0, stream>>>(t16, lb0t, hl, nullptr, NODES, RANK, HIDC);

  k_node_alpha<<<NODES, 128, 0, stream>>>(hb, hl, att_s0, att_d0, latt_s0, latt_d0,
                                          asb, adb, asl, adl, emb, eml, denb, denl, HIDC);
  k_edge_max<<<cdiv(NE2, TPB), TPB, 0, stream>>>(ei, asb, adb, asl, adl, emb, eml);
  k_edge_exp<<<cdiv(NE2, TPB), TPB, 0, stream>>>(ei, asb, adb, asl, adl, emb, eml,
                                                 exb, exl, denb, denl);
  k_init_bias<<<cdiv((long long)NODES * HIDC, TPB), TPB, 0, stream>>>(x1, b0, lbias0,
                                                                     NODES * HIDC, HIDC);
  k_edge_scatter<<<cdiv((long long)NE2 * (HIDC / 4), TPB), TPB, 0, stream>>>(
      ei, hb, hl, exb, exl, denb, denl, x1, x1, HIDC);

  // ================= layer 1 (C = 64) =================
  k_f32_to_bf16<<<cdiv((long long)NODES * HIDC, TPB), TPB, 0, stream>>>(x1, xb, NODES * HIDC);
  k_wmma_gemm<<<dim3(mblk, OUTC / 16), 128, 0, stream>>>(xb, w1t,  hb, nullptr, NODES, HIDC, OUTC);
  k_wmma_gemm<<<dim3(mblk, RANK / 16), 128, 0, stream>>>(xb, la1t, nullptr, t16, NODES, HIDC, RANK);
  k_wmma_gemm<<<dim3(mblk, OUTC / 16), 128, 0, stream>>>(t16, lb1t, hl, nullptr, NODES, RANK, OUTC);

  k_node_alpha<<<NODES, 128, 0, stream>>>(hb, hl, att_s1, att_d1, latt_s1, latt_d1,
                                          asb, adb, asl, adl, emb, eml, denb, denl, OUTC);
  k_edge_max<<<cdiv(NE2, TPB), TPB, 0, stream>>>(ei, asb, adb, asl, adl, emb, eml);
  k_edge_exp<<<cdiv(NE2, TPB), TPB, 0, stream>>>(ei, asb, adb, asl, adl, emb, eml,
                                                 exb, exl, denb, denl);

  float* e1 = out + (size_t)NODES * OUTC;        // second tuple element
  float* e2 = out + (size_t)2 * NODES * OUTC;    // third tuple element
  k_init_bias<<<cdiv((long long)NODES * OUTC, TPB), TPB, 0, stream>>>(e1, b1, nullptr,
                                                                     NODES * OUTC, OUTC);
  k_init_bias<<<cdiv((long long)NODES * OUTC, TPB), TPB, 0, stream>>>(e2, lbias1, nullptr,
                                                                     NODES * OUTC, OUTC);
  k_edge_scatter<<<cdiv((long long)NE2 * (OUTC / 4), TPB), TPB, 0, stream>>>(
      ei, hb, hl, exb, exl, denb, denl, e1, e2, OUTC);

  k_final_sum<<<cdiv((long long)NODES * OUTC, TPB), TPB, 0, stream>>>(e1, e2, out, NODES * OUTC);
}
